// ItemEmbeddingLayer_45165876085304
// MI455X (gfx1250) — matrix operand order repaired and checked by reading, not compiled, not run on hardware
//
#include <hip/hip_runtime.h>

// ---------------------------------------------------------------------------
// ItemEmbeddingLayer for MI455X (gfx1250):
//   out[8192,128] = Q[items]  @  E(=skill_embedding[user], 256x128)
// Strategy:
//   k1: convert E f32 -> f16 once, into "chunk" layout in d_ws:
//         ws[(c*128 + col)*8 + i] = E[c*8+i][col]   (c = 0..31, col = 0..127)
//       -> each 16B chunk = 8 consecutive K-values of one output column.
//   k2: per WG: TDM-copy the 64KB f16 B-panel into LDS (wave 0),
//       per wave: TDM gather-mode DMA of its 16 random Q rows into LDS
//       (padded rows, 260 floats, for bank rotation), then 8x8 = 64
//       v_wmma_f32_16x16x32_f16 per wave with f32 accumulation.
// ---------------------------------------------------------------------------

typedef _Float16 v16h __attribute__((ext_vector_type(16)));
typedef _Float16 v8h  __attribute__((ext_vector_type(8)));
typedef float    v8f  __attribute__((ext_vector_type(8)));
typedef unsigned int v4u __attribute__((ext_vector_type(4)));
typedef int      v8i  __attribute__((ext_vector_type(8)));
typedef int      v4i  __attribute__((ext_vector_type(4)));

static constexpr int SK     = 256;    // skills (reduction dim)
static constexpr int KD     = 128;    // hidden (output cols)
static constexpr int SEQ    = 8192;   // output rows
static constexpr int NITEMS = 4096;

static constexpr int QROW_PAD_F = 260;                 // floats per padded Q row in LDS
static constexpr int EPANEL_B   = SK * KD * 2;         // 65536 bytes (f16 B panel)
static constexpr int QSLICE_B   = 16 * QROW_PAD_F * 4; // 16640 bytes per wave
static constexpr int WAVES_PER_WG = 4;
static constexpr int LDS_BYTES  = EPANEL_B + WAVES_PER_WG * QSLICE_B; // 132096

#if __has_builtin(__builtin_amdgcn_tensor_load_to_lds)
#define HAVE_TDM 1
#else
#define HAVE_TDM 0
#endif

// ---------------------------------------------------------------------------
// Kernel 1: E (f32, 256x128) -> f16 chunk layout in d_ws (64KB).
// grid 16 x 256 threads: one thread per (chunk,col) unit.
// ---------------------------------------------------------------------------
__global__ void iel_cvtE_kernel(const int* __restrict__ userp,
                                const float* __restrict__ emb_all,
                                _Float16* __restrict__ ws) {
  const int t     = blockIdx.x * blockDim.x + threadIdx.x;  // 0..4095
  const int chunk = t >> 7;                                 // 0..31
  const int col   = t & 127;                                // 0..127
  const float* emb = emb_all + (size_t)(*userp) * (SK * KD);
  v8h h;
#pragma unroll
  for (int i = 0; i < 8; ++i)
    h[i] = (_Float16)emb[(chunk * 8 + i) * KD + col];       // coalesced across lanes
  *(v8h*)(ws + (size_t)t * 8) = h;
}

// ---------------------------------------------------------------------------
// Kernel 2: WMMA GEMM. 128 threads (4 waves), one 16-row M tile per wave.
// ---------------------------------------------------------------------------
__global__ void iel_wmma_kernel(const float* __restrict__ Qm,
                                const int* __restrict__ items,
                                const _Float16* __restrict__ wsE,
                                float* __restrict__ out) {
  extern __shared__ char smem[];
  _Float16* eT   = (_Float16*)smem;                 // 64KB f16 B panel (chunk layout)
  float*    qraw = (float*)(smem + EPANEL_B);       // 4 waves * 16 rows * 260 floats

  const int tid  = threadIdx.x;
  const int wave = tid >> 5;
  const int lane = tid & 31;
  const int hi   = lane >> 4;                       // 0: lanes 0-15, 1: lanes 16-31
  const int m    = lane & 15;                       // row within M tile / col within N tile
  const int mbase = (blockIdx.x * WAVES_PER_WG + wave) * 16;

  // each lane fetches the item index for row (lane&15) of its wave's M tile
  const int it = items[mbase + m];

#if HAVE_TDM
  {
    // ---- per-wave TDM gather: 16 Q rows (16-bit indices) -> padded LDS rows ----
    const unsigned ldsQ = (unsigned)(size_t)(qraw) + (unsigned)(wave * QSLICE_B);
    const unsigned long long ga = (unsigned long long)(size_t)Qm;
    v4u g0;
    g0[0] = 0x80000001u;                            // count=1, gather_mode=1, 16b idx
    g0[1] = ldsQ;                                   // lds_addr
    g0[2] = (unsigned)ga;                           // global_addr[31:0]
    g0[3] = ((unsigned)(ga >> 32) & 0x01FFFFFFu) | 0x80000000u; // addr[56:32] | type=2
    v8i g1;
    g1[0] = (2 << 16) | (1 << 20) | (7 << 22) | (3 << 25); // 4B elems, pad 4dw / 256dw
    g1[1] = (SK & 0xFFFF) << 16;                    // tensor_dim0 = 256
    g1[2] = (NITEMS & 0xFFFF) << 16;                // dim0 hi=0 | tensor_dim1 = 4096
    g1[3] = (SK & 0xFFFF) << 16;                    // dim1 hi=0 | tile_dim0 = 256
    g1[4] = 16;                                     // tile_dim1 = #valid indices
    g1[5] = SK;                                     // tensor_dim0_stride = 256
    g1[6] = 0;
    g1[7] = 0;
    v4i g2, g3;
#pragma unroll
    for (int i = 0; i < 4; ++i) {
      g2[i] = (__builtin_amdgcn_readlane(it, 2 * i) & 0xFFFF) |
              (__builtin_amdgcn_readlane(it, 2 * i + 1) << 16);
      g3[i] = (__builtin_amdgcn_readlane(it, 8 + 2 * i) & 0xFFFF) |
              (__builtin_amdgcn_readlane(it, 8 + 2 * i + 1) << 16);
    }
    v8i g4 = {0, 0, 0, 0, 0, 0, 0, 0};              // unused VADDR4 slot (zero)
    __builtin_amdgcn_tensor_load_to_lds(g0, g1, g2, g3, g4, 0);
  }
  if (wave == 0) {
    // ---- one TDM linear copy of the 64KB f16 B-panel into LDS ----
    const unsigned long long ga = (unsigned long long)(size_t)wsE;
    v4u g0;
    g0[0] = 1u;                                     // count=1
    g0[1] = (unsigned)(size_t)(eT);                 // lds_addr
    g0[2] = (unsigned)ga;
    g0[3] = ((unsigned)(ga >> 32) & 0x01FFFFFFu) | 0x80000000u;
    v8i g1;
    g1[0] = (2 << 16);                              // 4B elements
    g1[1] = (16384 & 0xFFFF) << 16;                 // tensor_dim0 = 16384 dwords
    g1[2] = (1 << 16);                              // dim0 hi=0 | tensor_dim1 = 1
    g1[3] = (16384 & 0xFFFF) << 16;                 // tile_dim0 = 16384
    g1[4] = 1;                                      // tile_dim1 = 1
    g1[5] = 16384;                                  // tensor_dim0_stride
    g1[6] = 0;
    g1[7] = 0;
    v4i z = {0, 0, 0, 0};
    v8i g4 = {0, 0, 0, 0, 0, 0, 0, 0};              // unused VADDR4 slot (zero)
    __builtin_amdgcn_tensor_load_to_lds(g0, g1, z, z, g4, 0);
  }
#if __has_builtin(__builtin_amdgcn_s_wait_tensorcnt)
  __builtin_amdgcn_s_wait_tensorcnt(0);
#else
  asm volatile("s_wait_tensorcnt 0x0" ::: "memory");
#endif
#else  // !HAVE_TDM fallback: plain cooperative copies
  {
    float* qw = qraw + wave * (16 * QROW_PAD_F);
#pragma unroll
    for (int row = 0; row < 16; ++row) {
      const int ritem = __builtin_amdgcn_readlane(it, row);
      for (int off4 = lane; off4 < 64; off4 += 32) {
        float4 v = *(const float4*)(Qm + (size_t)ritem * SK + off4 * 4);
        *(float4*)(qw + row * QROW_PAD_F + off4 * 4) = v;
      }
    }
    for (int j = tid; j < SK * KD / 8; j += blockDim.x)
      ((v8h*)eT)[j] = ((const v8h*)wsE)[j];
  }
#endif
  __syncthreads();   // B panel visible to all waves

  // ------------------------- WMMA main loop --------------------------------
  const int koffA = hi * 8;                 // A K-offset per lane group (ISA 7.12.2)
  const float* qrow = qraw + wave * (16 * QROW_PAD_F) + m * QROW_PAD_F;

  v8f acc[8] = {};                          // 8 N-tiles of 16x16 f32 accumulators

#pragma unroll
  for (int ks = 0; ks < 8; ++ks) {
    const int k0 = ks * 32;
    // A fragment: Q row m, f32 -> f16 (exact: Q is 0/1)
    float4 qa = *(const float4*)(qrow + k0 + koffA);
    float4 qb = *(const float4*)(qrow + k0 + koffA + 4);
    float4 qc = *(const float4*)(qrow + k0 + 16 + koffA);
    float4 qd = *(const float4*)(qrow + k0 + 16 + koffA + 4);
    union { v16h v; _Float16 e[16]; } A;
    A.e[0]  = (_Float16)qa.x; A.e[1]  = (_Float16)qa.y;
    A.e[2]  = (_Float16)qa.z; A.e[3]  = (_Float16)qa.w;
    A.e[4]  = (_Float16)qb.x; A.e[5]  = (_Float16)qb.y;
    A.e[6]  = (_Float16)qb.z; A.e[7]  = (_Float16)qb.w;
    A.e[8]  = (_Float16)qc.x; A.e[9]  = (_Float16)qc.y;
    A.e[10] = (_Float16)qc.z; A.e[11] = (_Float16)qc.w;
    A.e[12] = (_Float16)qd.x; A.e[13] = (_Float16)qd.y;
    A.e[14] = (_Float16)qd.z; A.e[15] = (_Float16)qd.w;

    const int c0 = ks * 4 + hi;             // B chunk index for this lane group
#pragma unroll
    for (int n = 0; n < 8; ++n) {
      const int col = n * 16 + m;
      union { v16h v; v8h p[2]; } B;
      B.p[0] = *(const v8h*)(eT + (size_t)(c0 * 128 + col) * 8);
      B.p[1] = *(const v8h*)(eT + (size_t)((c0 + 2) * 128 + col) * 8);
      acc[n] = __builtin_amdgcn_wmma_f32_16x16x32_f16(
          /*neg_a=*/false, A.v, /*neg_b=*/false, B.v,
          /*c_mod=*/(short)0, acc[n], /*reuse_a=*/false, /*reuse_b=*/false);
    }
  }

  // ---- store: row = mbase + r + 8*hi, col = 16n + m (ISA C/D layout) ----
#pragma unroll
  for (int n = 0; n < 8; ++n) {
    const int col = n * 16 + m;
#pragma unroll
    for (int r = 0; r < 8; ++r) {
      const int row = mbase + r + hi * 8;
      out[(size_t)row * KD + col] = acc[n][r];
    }
  }
}

// ---------------------------------------------------------------------------
extern "C" void kernel_launch(void* const* d_in, const int* in_sizes, int n_in,
                              void* d_out, int out_size, void* d_ws, size_t ws_size,
                              hipStream_t stream) {
  const int*   userp = (const int*)d_in[0];
  const float* Qm    = (const float*)d_in[1];
  const int*   items = (const int*)d_in[2];
  const float* emb   = (const float*)d_in[3];
  float*       out   = (float*)d_out;
  _Float16*    ws    = (_Float16*)d_ws;     // needs 64KB scratch

  // Kernel 1: 4096 (chunk,col) units, 16 blocks x 256 threads
  iel_cvtE_kernel<<<dim3(16), dim3(256), 0, stream>>>(userp, emb, ws);

  // Kernel 2: 512 M-tiles / 4 waves per WG -> 128 WGs of 128 threads
  iel_wmma_kernel<<<dim3(SEQ / (16 * WAVES_PER_WG)), dim3(32 * WAVES_PER_WG),
                    LDS_BYTES, stream>>>(Qm, items, ws, out);
}
